// T5Decoder__29334626632463
// MI455X (gfx1250) — compile-verified, matrix operand order checked by reading, not run on hardware
//
#include <hip/hip_runtime.h>
#include <hip/hip_bf16.h>

// ---------------------------------------------------------------------------
// T5 decoder layer on MI455X (gfx1250, wave32, WMMA).
//  - all GEMMs + attention cores: v_wmma_f32_16x16x32_f16
//  - weights pre-transposed to [N,K] f16 so both WMMA operands load as
//    contiguous global_load_b128 "A-style" fragments
//  - GEMM wave tile 32x64: 8 WMMA per 12 b128 loads, loads hoisted so the
//    compiler emits one load clause + one wait + a WMMA burst per k-step
//  - flash attention (online softmax), scores never hit memory
// ---------------------------------------------------------------------------

typedef __attribute__((ext_vector_type(16))) _Float16 v16h;
typedef __attribute__((ext_vector_type(8)))  _Float16 v8h;
typedef __attribute__((ext_vector_type(8)))  float    v8f;

#define T5_B    4
#define T5_LD   1024
#define T5_LE   1024
#define T5_D    1024
#define T5_H    16
#define T5_DKV  64
#define T5_DFF  4096
#define T5_NEG  (-1e9f)

__device__ __forceinline__ v8f wmma_f16(v16h a, v16h b, v8f c) {
  // (neg_a, A, neg_b, B, c_mod, C, reuse_a, reuse_b)
  return __builtin_amdgcn_wmma_f32_16x16x32_f16(false, a, false, b, (short)0, c,
                                                false, false);
}

// Load a 16x32 f16 "A-style" fragment from row-major [rows, ld] memory.
// ISA layout: lanes 0-15 hold M=lane, halves {K0..7, K16..23};
//             lanes 16-31 hold M=lane-16, halves {K8..15, K24..31}.
// Both halves are contiguous 16B runs -> two b128 loads per lane.
__device__ __forceinline__ v16h load_frag(const _Float16* base, int ld,
                                          int row0, int col0, int lane) {
  const int hl = lane >> 4;
  const int r  = row0 + (lane & 15);
  const _Float16* p = base + (size_t)r * ld + col0 + hl * 8;
  union { v16h v; v8h h[2]; } u;
  u.h[0] = *(const v8h*)(p);
  u.h[1] = *(const v8h*)(p + 16);
  return u.v;
}

// ---------------------------------------------------------------------------
// Conversions
// ---------------------------------------------------------------------------
__global__ void cvt16_kernel(const float* __restrict__ in,
                             _Float16* __restrict__ out, int n) {
  int i = blockIdx.x * 256 + threadIdx.x;
  if (i < n) out[i] = (_Float16)in[i];
}

// W [K,N] f32 -> Wt [N,K] f16
__global__ void cvtT_kernel(const float* __restrict__ in,
                            _Float16* __restrict__ out, int K, int N) {
  int i = blockIdx.x * 256 + threadIdx.x;
  if (i >= K * N) return;
  int k = i / N, n = i - k * N;
  out[(size_t)n * K + k] = (_Float16)in[i];
}

// T5 relative-position bias table: bd[h][delta] for delta = q-k in [0,1023]
__global__ void relbias_kernel(const float* __restrict__ rel_bias, // [32,16]
                               float* __restrict__ bd) {           // [16,1024]
  int idx = blockIdx.x * 256 + threadIdx.x;
  if (idx >= T5_H * 1024) return;
  int h = idx >> 10, n = idx & 1023;
  int bucket;
  if (n < 16) {
    bucket = n;
  } else {
    float nf = (float)n;
    int vl = 16 + (int)(__logf(nf * (1.0f / 16.0f)) * (16.0f / __logf(8.0f)));
    bucket = vl < 31 ? vl : 31;
  }
  bd[idx] = rel_bias[bucket * T5_H + h];
}

// ---------------------------------------------------------------------------
// RMS norm: one block per row of [rows, 1024]
// ---------------------------------------------------------------------------
__global__ __launch_bounds__(256)
void rmsnorm_kernel(const float* __restrict__ X, const float* __restrict__ w,
                    _Float16* __restrict__ y16, float* __restrict__ y32) {
  __shared__ float red[256];
  const int row = blockIdx.x;
  const float* x = X + (size_t)row * T5_D;
  float ss = 0.f;
  for (int i = threadIdx.x; i < T5_D; i += 256) { float v = x[i]; ss += v * v; }
  red[threadIdx.x] = ss;
  __syncthreads();
  for (int s = 128; s > 0; s >>= 1) {
    if (threadIdx.x < s) red[threadIdx.x] += red[threadIdx.x + s];
    __syncthreads();
  }
  float rs = rsqrtf(red[0] * (1.0f / (float)T5_D) + 1e-6f);
  for (int i = threadIdx.x; i < T5_D; i += 256) {
    float v = x[i] * rs * w[i];
    if (y16) y16[(size_t)row * T5_D + i] = (_Float16)v;
    if (y32) y32[(size_t)row * T5_D + i] = v;
  }
}

// ---------------------------------------------------------------------------
// WMMA GEMM: C[M,N] = A[M,K] @ Bt[N,K]^T
// Workgroup = 4 waves; wave tile = 32 rows x 64 cols:
//   2 A-fragments x 4 B-fragments -> 8 WMMA per k-step (12 b128 loads).
// All fragments of a k-step loaded before the WMMA burst; next k-step
// prefetched (global_prefetch_b8) to prime L2/WGP$.
// ---------------------------------------------------------------------------
#define GMODE_RESID  0   // f32 residual: Cres[m*N+n] += c
#define GMODE_QKV    1   // f16 [B,H,L,64]
#define GMODE_VT     2   // f16 [B,H,64,L] (pre-transposed V)
#define GMODE_RELU16 3   // f16 relu, row-major [M,N]

__global__ __launch_bounds__(128)
void gemm_wmma(const _Float16* __restrict__ A, const _Float16* __restrict__ Bt,
               float* __restrict__ Cres, _Float16* __restrict__ C16,
               int M, int N, int K, int mode, int Lrows) {
  const int lane = threadIdx.x & 31;
  const int wave = threadIdx.x >> 5;
  const int hl   = lane >> 4;
  const int row0 = blockIdx.x * 128 + wave * 32;
  const int n0   = blockIdx.y * 64;

  const v8f zero8 = {0.f, 0.f, 0.f, 0.f, 0.f, 0.f, 0.f, 0.f};
  v8f acc[8];                      // [t] rows row0..+15, [4+t] rows row0+16..+31
#pragma unroll
  for (int t = 0; t < 8; ++t) acc[t] = zero8;

  for (int kk = 0; kk < K; kk += 32) {
    // hoist all fragment loads -> single clause + single wait before WMMAs
    v16h a0 = load_frag(A, K, row0,      kk, lane);
    v16h a1 = load_frag(A, K, row0 + 16, kk, lane);
    v16h b0 = load_frag(Bt, K, n0,      kk, lane);
    v16h b1 = load_frag(Bt, K, n0 + 16, kk, lane);
    v16h b2 = load_frag(Bt, K, n0 + 32, kk, lane);
    v16h b3 = load_frag(Bt, K, n0 + 48, kk, lane);

    if (kk + 32 < K) {   // prime next k-step (lowers to global_prefetch_b8)
      __builtin_prefetch(A  + (size_t)(row0 + (lane & 15)) * K + kk + 32, 0, 3);
      __builtin_prefetch(Bt + (size_t)(n0 + (lane & 15)) * K + kk + 32, 0, 3);
    }

    acc[0] = wmma_f16(a0, b0, acc[0]);
    acc[1] = wmma_f16(a0, b1, acc[1]);
    acc[2] = wmma_f16(a0, b2, acc[2]);
    acc[3] = wmma_f16(a0, b3, acc[3]);
    acc[4] = wmma_f16(a1, b0, acc[4]);
    acc[5] = wmma_f16(a1, b1, acc[5]);
    acc[6] = wmma_f16(a1, b2, acc[6]);
    acc[7] = wmma_f16(a1, b3, acc[7]);
  }

#pragma unroll
  for (int g = 0; g < 2; ++g) {
#pragma unroll
    for (int t = 0; t < 4; ++t) {
#pragma unroll
      for (int r = 0; r < 8; ++r) {
        // C-fragment layout: lane half picks row group, VGPR r picks row
        int m = row0 + g * 16 + r + hl * 8;
        int n = n0 + t * 16 + (lane & 15);
        float c = acc[g * 4 + t][r];
        if (mode == GMODE_RESID) {
          Cres[(size_t)m * N + n] += c;
        } else if (mode == GMODE_RELU16) {
          C16[(size_t)m * N + n] = (_Float16)fmaxf(c, 0.0f);
        } else {
          int bb = m / Lrows, l = m - bb * Lrows;
          int hh = n >> 6, d = n & 63;
          size_t idx;
          if (mode == GMODE_QKV)
            idx = (((size_t)bb * T5_H + hh) * Lrows + l) * T5_DKV + d;
          else
            idx = (((size_t)bb * T5_H + hh) * T5_DKV + d) * (size_t)Lrows + l;
          C16[idx] = (_Float16)c;
        }
      }
    }
  }
}

// ---------------------------------------------------------------------------
// Flash attention (WMMA). Per wave: 16 queries; key blocks of 32.
// Q,K: [B,H,L,64] f16.  Vt: [B,H,64,L] f16.  Out: [B,LQ,1024] f16.
// Online softmax; P transposed C->A fragment via 1KB/wave LDS bounce
// (same-wave DS ordering, no barrier => safe with causal early exit).
// ---------------------------------------------------------------------------
__global__ __launch_bounds__(128)
void attn_wmma_kernel(const _Float16* __restrict__ Q,
                      const _Float16* __restrict__ Km,
                      const _Float16* __restrict__ Vt,
                      const float* __restrict__ bias_delta, // [16,1024]
                      const int* __restrict__ mask,         // [B,LK]
                      _Float16* __restrict__ Out,
                      int LQ, int LK, int causal, int use_rel) {
  __shared__ __align__(16) _Float16 plds[4][16][32];
  const int lane = threadIdx.x & 31;
  const int wave = threadIdx.x >> 5;
  const int hl   = lane >> 4;
  const int col  = lane & 15;
  const int bh = blockIdx.y;
  const int b  = bh >> 4;
  const int h  = bh & 15;
  const int q0 = blockIdx.x * 64 + wave * 16;

  const _Float16* Qb = Q  + (size_t)bh * LQ * T5_DKV;
  const _Float16* Kb = Km + (size_t)bh * LK * T5_DKV;
  const _Float16* Vb = Vt + (size_t)bh * T5_DKV * LK;
  const int* mk = mask + (size_t)b * LK;
  const float* bd = bias_delta + h * 1024;

  v16h qa0 = load_frag(Qb, T5_DKV, q0, 0, lane);
  v16h qa1 = load_frag(Qb, T5_DKV, q0, 32, lane);

  const v8f zero8 = {0.f, 0.f, 0.f, 0.f, 0.f, 0.f, 0.f, 0.f};
  v8f o[4];
  float mrow[8], lrow[8];
#pragma unroll
  for (int t = 0; t < 4; ++t) o[t] = zero8;
#pragma unroll
  for (int r = 0; r < 8; ++r) { mrow[r] = -1e30f; lrow[r] = 0.f; }

  for (int kb = 0; kb < LK; kb += 32) {
    if (causal && kb >= q0 + 16) break;  // whole block above the diagonal

    v16h k00 = load_frag(Kb, T5_DKV, kb,      0,  lane);
    v16h k01 = load_frag(Kb, T5_DKV, kb,      32, lane);
    v16h k10 = load_frag(Kb, T5_DKV, kb + 16, 0,  lane);
    v16h k11 = load_frag(Kb, T5_DKV, kb + 16, 32, lane);

    v8f s0 = wmma_f16(qa0, k00, zero8); s0 = wmma_f16(qa1, k01, s0);
    v8f s1 = wmma_f16(qa0, k10, zero8); s1 = wmma_f16(qa1, k11, s1);

#pragma unroll
    for (int r = 0; r < 8; ++r) {
      int qr  = q0 + r + hl * 8;
      int k0c = kb + col;
      int k1c = kb + 16 + col;
      float a0 = s0[r], a1 = s1[r];
      if (use_rel) {
        if (k0c <= qr) a0 += bd[qr - k0c];
        if (k1c <= qr) a1 += bd[qr - k1c];
      }
      if (causal) {
        if (k0c > qr) a0 = T5_NEG;
        if (k1c > qr) a1 = T5_NEG;
      }
      if (mk[k0c] == 0) a0 = T5_NEG;
      if (mk[k1c] == 0) a1 = T5_NEG;

      // row max over the 16 lanes of this half
      float v = fmaxf(a0, a1);
      v = fmaxf(v, __shfl_xor(v, 1, 16));
      v = fmaxf(v, __shfl_xor(v, 2, 16));
      v = fmaxf(v, __shfl_xor(v, 4, 16));
      v = fmaxf(v, __shfl_xor(v, 8, 16));
      float mnew  = fmaxf(mrow[r], v);
      float scale = __expf(mrow[r] - mnew);
      mrow[r] = mnew;
      float e0 = __expf(a0 - mnew);
      float e1 = __expf(a1 - mnew);
      float sum = e0 + e1;
      sum += __shfl_xor(sum, 1, 16);
      sum += __shfl_xor(sum, 2, 16);
      sum += __shfl_xor(sum, 4, 16);
      sum += __shfl_xor(sum, 8, 16);
      lrow[r] = lrow[r] * scale + sum;
#pragma unroll
      for (int t = 0; t < 4; ++t) o[t][r] *= scale;

      int prow = r + hl * 8;
      plds[wave][prow][col]      = (_Float16)e0;
      plds[wave][prow][col + 16] = (_Float16)e1;
    }

    // C-fragment -> A-fragment transpose through per-wave LDS
    v16h pa = load_frag(&plds[wave][0][0], 32, 0, 0, lane);
    v16h vb0 = load_frag(Vb, LK, 0,  kb, lane);
    v16h vb1 = load_frag(Vb, LK, 16, kb, lane);
    v16h vb2 = load_frag(Vb, LK, 32, kb, lane);
    v16h vb3 = load_frag(Vb, LK, 48, kb, lane);
    o[0] = wmma_f16(pa, vb0, o[0]);
    o[1] = wmma_f16(pa, vb1, o[1]);
    o[2] = wmma_f16(pa, vb2, o[2]);
    o[3] = wmma_f16(pa, vb3, o[3]);
  }

#pragma unroll
  for (int t = 0; t < 4; ++t) {
#pragma unroll
    for (int r = 0; r < 8; ++r) {
      int qr = q0 + r + hl * 8;
      float val = o[t][r] / lrow[r];
      Out[((size_t)b * LQ + qr) * (T5_H * T5_DKV) + h * T5_DKV + t * 16 + col] =
          (_Float16)val;
    }
  }
}

// ---------------------------------------------------------------------------
// Orchestration
// ---------------------------------------------------------------------------
extern "C" void kernel_launch(void* const* d_in, const int* in_sizes, int n_in,
                              void* d_out, int out_size, void* d_ws,
                              size_t ws_size, hipStream_t stream) {
  (void)in_sizes; (void)n_in; (void)out_size; (void)ws_size;

  const float* enc   = (const float*)d_in[0];
  const float* hid   = (const float*)d_in[1];
  const float* ln1   = (const float*)d_in[2];
  const float* saq   = (const float*)d_in[3];
  const float* sak   = (const float*)d_in[4];
  const float* sav   = (const float*)d_in[5];
  const float* sao   = (const float*)d_in[6];
  const float* relb  = (const float*)d_in[7];
  const float* ln2   = (const float*)d_in[8];
  const float* caq   = (const float*)d_in[9];
  const float* cak   = (const float*)d_in[10];
  const float* cav   = (const float*)d_in[11];
  const float* cao   = (const float*)d_in[12];
  const float* ln3   = (const float*)d_in[13];
  const float* wi    = (const float*)d_in[14];
  const float* wo    = (const float*)d_in[15];
  const float* lnf   = (const float*)d_in[16];
  const int* encmask = (const int*)d_in[17];
  const int* decmask = (const int*)d_in[18];
  float* out = (float*)d_out;

  const size_t M = (size_t)T5_B * T5_LD;        // 4096 decoder rows
  char* ws = (char*)d_ws;
  size_t off = 0;
  auto alloc = [&](size_t bytes) -> void* {
    void* p = ws + off;
    off = (off + bytes + 255) & ~(size_t)255;
    return p;
  };

  float*    h_cur = (float*)   alloc(M * T5_D * 4);            // 16 MB residual
  _Float16* xn    = (_Float16*)alloc(M * T5_D * 2);            // 8 MB
  _Float16* saqT  = (_Float16*)alloc((size_t)T5_D * T5_D * 2);
  _Float16* sakT  = (_Float16*)alloc((size_t)T5_D * T5_D * 2);
  _Float16* savT  = (_Float16*)alloc((size_t)T5_D * T5_D * 2);
  _Float16* saoT  = (_Float16*)alloc((size_t)T5_D * T5_D * 2);
  _Float16* caqT  = (_Float16*)alloc((size_t)T5_D * T5_D * 2);
  _Float16* cakT  = (_Float16*)alloc((size_t)T5_D * T5_D * 2);
  _Float16* cavT  = (_Float16*)alloc((size_t)T5_D * T5_D * 2);
  _Float16* caoT  = (_Float16*)alloc((size_t)T5_D * T5_D * 2);
  _Float16* wiT   = (_Float16*)alloc((size_t)T5_DFF * T5_D * 2);
  _Float16* woT   = (_Float16*)alloc((size_t)T5_D * T5_DFF * 2);
  _Float16* enc16 = (_Float16*)alloc((size_t)T5_B * T5_LE * T5_D * 2);
  _Float16* qbuf  = (_Float16*)alloc(M * T5_D * 2);            // [B,H,L,64]
  _Float16* kbuf  = (_Float16*)alloc(M * T5_D * 2);            // [B,H,L,64]
  _Float16* vtbuf = (_Float16*)alloc(M * T5_D * 2);            // [B,H,64,L]
  _Float16* aout  = (_Float16*)alloc(M * T5_D * 2);            // [B*L, 1024]
  _Float16* ffn   = (_Float16*)alloc(M * T5_DFF * 2);          // 32 MB
  float*    bd    = (float*)   alloc((size_t)T5_H * 1024 * 4); // rel bias table

  // residual stream starts as hidden_states (never mutate inputs)
  hipMemcpyAsync(h_cur, hid, M * T5_D * 4, hipMemcpyDeviceToDevice, stream);

  // --- weight prep (f32 -> f16, transposed to [N,K]) ---
  const int DD = T5_D * T5_D;  // 1M
  cvtT_kernel<<<(DD + 255) / 256, 256, 0, stream>>>(saq, saqT, T5_D, T5_D);
  cvtT_kernel<<<(DD + 255) / 256, 256, 0, stream>>>(sak, sakT, T5_D, T5_D);
  cvtT_kernel<<<(DD + 255) / 256, 256, 0, stream>>>(sav, savT, T5_D, T5_D);
  cvtT_kernel<<<(DD + 255) / 256, 256, 0, stream>>>(sao, saoT, T5_D, T5_D);
  cvtT_kernel<<<(DD + 255) / 256, 256, 0, stream>>>(caq, caqT, T5_D, T5_D);
  cvtT_kernel<<<(DD + 255) / 256, 256, 0, stream>>>(cak, cakT, T5_D, T5_D);
  cvtT_kernel<<<(DD + 255) / 256, 256, 0, stream>>>(cav, cavT, T5_D, T5_D);
  cvtT_kernel<<<(DD + 255) / 256, 256, 0, stream>>>(cao, caoT, T5_D, T5_D);
  cvtT_kernel<<<(T5_D * T5_DFF + 255) / 256, 256, 0, stream>>>(wi, wiT, T5_D, T5_DFF);
  cvtT_kernel<<<(T5_DFF * T5_D + 255) / 256, 256, 0, stream>>>(wo, woT, T5_DFF, T5_D);
  cvt16_kernel<<<(int)((M * T5_D + 255) / 256), 256, 0, stream>>>(enc, enc16,
                                                                 (int)(M * T5_D));
  relbias_kernel<<<(T5_H * 1024 + 255) / 256, 256, 0, stream>>>(relb, bd);

  dim3 gblk(128);
  dim3 ggrid_d(M / 128, T5_D / 64);    // N = 1024
  dim3 ggrid_ff(M / 128, T5_DFF / 64); // N = 4096
  dim3 agrid(T5_LD / 64, T5_B * T5_H);
  dim3 ablk(128);

  // ---- self attention ----
  rmsnorm_kernel<<<(int)M, 256, 0, stream>>>(h_cur, ln1, xn, nullptr);
  gemm_wmma<<<ggrid_d, gblk, 0, stream>>>(xn, saqT, nullptr, qbuf,
                                          (int)M, T5_D, T5_D, GMODE_QKV, T5_LD);
  gemm_wmma<<<ggrid_d, gblk, 0, stream>>>(xn, sakT, nullptr, kbuf,
                                          (int)M, T5_D, T5_D, GMODE_QKV, T5_LD);
  gemm_wmma<<<ggrid_d, gblk, 0, stream>>>(xn, savT, nullptr, vtbuf,
                                          (int)M, T5_D, T5_D, GMODE_VT, T5_LD);
  attn_wmma_kernel<<<agrid, ablk, 0, stream>>>(qbuf, kbuf, vtbuf, bd, decmask,
                                               aout, T5_LD, T5_LD,
                                               /*causal=*/1, /*use_rel=*/1);
  gemm_wmma<<<ggrid_d, gblk, 0, stream>>>(aout, saoT, h_cur, nullptr,
                                          (int)M, T5_D, T5_D, GMODE_RESID, T5_LD);

  // ---- cross attention ----
  rmsnorm_kernel<<<(int)M, 256, 0, stream>>>(h_cur, ln2, xn, nullptr);
  gemm_wmma<<<ggrid_d, gblk, 0, stream>>>(xn, caqT, nullptr, qbuf,
                                          (int)M, T5_D, T5_D, GMODE_QKV, T5_LD);
  gemm_wmma<<<ggrid_d, gblk, 0, stream>>>(enc16, cakT, nullptr, kbuf,
                                          (int)M, T5_D, T5_D, GMODE_QKV, T5_LE);
  gemm_wmma<<<ggrid_d, gblk, 0, stream>>>(enc16, cavT, nullptr, vtbuf,
                                          (int)M, T5_D, T5_D, GMODE_VT, T5_LE);
  attn_wmma_kernel<<<agrid, ablk, 0, stream>>>(qbuf, kbuf, vtbuf, bd, encmask,
                                               aout, T5_LD, T5_LE,
                                               /*causal=*/0, /*use_rel=*/0);
  gemm_wmma<<<ggrid_d, gblk, 0, stream>>>(aout, caoT, h_cur, nullptr,
                                          (int)M, T5_D, T5_D, GMODE_RESID, T5_LD);

  // ---- FFN ----
  rmsnorm_kernel<<<(int)M, 256, 0, stream>>>(h_cur, ln3, xn, nullptr);
  gemm_wmma<<<ggrid_ff, gblk, 0, stream>>>(xn, wiT, nullptr, ffn,
                                           (int)M, T5_DFF, T5_D, GMODE_RELU16, T5_LD);
  gemm_wmma<<<ggrid_d, gblk, 0, stream>>>(ffn, woT, h_cur, nullptr,
                                          (int)M, T5_D, T5_DFF, GMODE_RESID, T5_LD);

  // ---- final norm -> f32 output ----
  rmsnorm_kernel<<<(int)M, 256, 0, stream>>>(h_cur, lnf, nullptr, out);
}
